// Recur_self_attn_75505525063777
// MI455X (gfx1250) — matrix operand order repaired and checked by reading, not compiled, not run on hardware
//
#include <hip/hip_runtime.h>
#include <hip/hip_bf16.h>
#include <math.h>

// ---------------- CDNA5 WMMA types ----------------
typedef __attribute__((ext_vector_type(16))) __bf16 v16bf;
typedef __attribute__((ext_vector_type(8)))  float  v8f;

#define WMMA_BF16(A, B, C) \
  __builtin_amdgcn_wmma_f32_16x16x32_bf16(false, (A), false, (B), (short)0, (C), false, false)

__device__ __forceinline__ __bf16 f2bf(float f) { return (__bf16)f; }
__device__ __forceinline__ float  rcpf(float x) { return __builtin_amdgcn_rcpf(x); }

#define LOG2E 1.4426950408889634f

// C/D 16x16 f32: value v of a lane sits at row (v or v+8), col = lane%16.
__device__ __forceinline__ int c_row(int lane, int v) {
  return (lane < 16) ? v : (v + 8);
}
// Inverse A-frag map (16x32 16-bit): element (row M, k) lives at
//   lane = M + ((k&8)?16:0),  half = (k&7) + ((k&16)?8:0)
__device__ __forceinline__ int a_lane(int M, int k) { return M + ((k & 8) ? 16 : 0); }
__device__ __forceinline__ int a_half(int k)        { return (k & 7) + ((k & 16) ? 8 : 0); }
// Inverse B-frag map (32x16 16-bit): element (k, col N) lives at
//   lane = N + ((k&8)?16:0),  half = (k&7) + ((k&16)?8:0)
__device__ __forceinline__ int b_lane(int N, int k) { return N + ((k & 8) ? 16 : 0); }
__device__ __forceinline__ int b_half(int k)        { return (k & 7) + ((k & 16) ? 8 : 0); }

#define BB     4
#define NN     2048
#define DIM    64
#define HEADS  8
#define INNER  512
#define TOTROW (BB * NN)       // 8192
#define NC3    (3 * INNER)     // 1536
#define EPS    1e-5f

// Load one 16x32 A-fragment from an f32 row (two contiguous 8-float K groups/lane).
__device__ __forceinline__ v16bf load_a_f32(const float* row, int kc, int lane) {
  const float* p = row + kc * 32 + ((lane < 16) ? 0 : 8);
  float4 a = *(const float4*)(p);
  float4 b = *(const float4*)(p + 4);
  float4 c = *(const float4*)(p + 16);
  float4 d = *(const float4*)(p + 20);
  v16bf o;
  o[0] = f2bf(a.x);  o[1] = f2bf(a.y);  o[2]  = f2bf(a.z);  o[3]  = f2bf(a.w);
  o[4] = f2bf(b.x);  o[5] = f2bf(b.y);  o[6]  = f2bf(b.z);  o[7]  = f2bf(b.w);
  o[8] = f2bf(c.x);  o[9] = f2bf(c.y);  o[10] = f2bf(c.z);  o[11] = f2bf(c.w);
  o[12] = f2bf(d.x); o[13] = f2bf(d.y); o[14] = f2bf(d.z);  o[15] = f2bf(d.w);
  return o;
}

// =====================================================================
// Kernel 0: pack weights into WMMA B-fragment layout (one-time, tiny).
// WcP/WpP: [ntile(96)][kc(2)][lane(32)][h(16)],  WjP: [kc(16)][dt(4)][lane][h]
// =====================================================================
__global__ __launch_bounds__(256) void pack_w_kernel(
    const float* __restrict__ Wc, const float* __restrict__ Wp,
    const float* __restrict__ Wj,
    __bf16* __restrict__ WcP, __bf16* __restrict__ WpP, __bf16* __restrict__ WjP)
{
  const int idx = blockIdx.x * 256 + threadIdx.x;
  if (idx < 96 * 2 * 32 * 16) {
    int h = idx & 15, lane = (idx >> 4) & 31, kc = (idx >> 9) & 1, nt = idx >> 10;
    int k = kc * 32 + ((lane < 16) ? 0 : 8) + (h & 7) + ((h & 8) ? 16 : 0);
    int c = nt * 16 + (lane & 15);
    WcP[idx] = f2bf(Wc[(size_t)k * NC3 + c]);
    WpP[idx] = f2bf(Wp[(size_t)k * NC3 + c]);
  }
  if (idx < 16 * 4 * 32 * 16) {
    int h = idx & 15, lane = (idx >> 4) & 31, dt = (idx >> 9) & 3, kc = idx >> 11;
    int k = kc * 32 + ((lane < 16) ? 0 : 8) + (h & 7) + ((h & 8) ? 16 : 0);
    int c = dt * 16 + (lane & 15);
    WjP[idx] = f2bf(Wj[(size_t)k * DIM + c]);
  }
}

// =====================================================================
// Kernel 1: gated QKV.  One wave per 16x16 output tile.  Writes Q/K/V
// pre-swizzled into WMMA fragment layout:
//   Qp: [bh][strip(128)][kc(2)][lane][h]   (A-fragments of Q rows)
//   Kp: [bh][jt(128)][kc(2)][lane][h]      (B-fragments of K^T)
//   Vp: [bh][kt(64)][dt(4)][lane][h]       (B-fragments of V)
// =====================================================================
__global__ __launch_bounds__(256) void qkv_gate_kernel(
    const float* __restrict__ cur, const float* __restrict__ pre,
    const v16bf* __restrict__ WcP, const v16bf* __restrict__ WpP,
    const float* __restrict__ bc_, const float* __restrict__ bp_,
    const float* __restrict__ alpha,
    __bf16* __restrict__ Qp, __bf16* __restrict__ Kp, __bf16* __restrict__ Vp)
{
  const int tid  = threadIdx.x;
  const int wave = tid >> 5;
  const int lane = tid & 31;
  const int col  = lane & 15;

  const int gwave = blockIdx.x * 8 + wave;      // 512*96 tiles
  const int mtile = gwave / (NC3 / 16);
  const int ntile = gwave % (NC3 / 16);
  const int m0 = mtile * 16;
  const int n0 = ntile * 16;

  const float* xc = cur + (size_t)(m0 + col) * DIM;
  const float* xp = pre + (size_t)(m0 + col) * DIM;

  v16bf ac0 = load_a_f32(xc, 0, lane), ac1 = load_a_f32(xc, 1, lane);
  v16bf ap0 = load_a_f32(xp, 0, lane), ap1 = load_a_f32(xp, 1, lane);
  v16bf wc0 = WcP[(size_t)ntile * 64 + lane];
  v16bf wc1 = WcP[(size_t)ntile * 64 + 32 + lane];
  v16bf wp0 = WpP[(size_t)ntile * 64 + lane];
  v16bf wp1 = WpP[(size_t)ntile * 64 + 32 + lane];

  v8f accC = {}, accP = {};
  accC = WMMA_BF16(ac0, wc0, accC);
  accC = WMMA_BF16(ac1, wc1, accC);
  accP = WMMA_BF16(ap0, wp0, accP);
  accP = WMMA_BF16(ap1, wp1, accP);

  const int c   = n0 + col;
  const int qkv = c / INNER;                  // uniform across the wave's tile
  const int idx = c - qkv * INNER;
  const int hh  = idx >> 6;
  const int dd  = idx & 63;
  const float g  = rcpf(1.0f + exp2f(-alpha[qkv * INNER + idx] * LOG2E));
  const float bc = bc_[c];
  const float bp = bp_[c];

  for (int v = 0; v < 8; ++v) {
    const int r  = m0 + c_row(lane, v);
    const int bb = r >> 11, ii = r & (NN - 1);
    const int bh = bb * HEADS + hh;
    const float val = g * (accC[v] + bc) + (1.0f - g) * (accP[v] + bp);
    const __bf16 bv = f2bf(val);
    if (qkv == 0) {            // Q: A-frag of row ii, k = dd
      int strip = ii >> 4, irow = ii & 15, kc = dd >> 5, kl = dd & 31;
      Qp[(((size_t)bh * 128 + strip) * 2 + kc) * 512 + a_lane(irow, kl) * 16 + a_half(kl)] = bv;
    } else if (qkv == 1) {     // K: B-frag column = token ii, k = dd
      int jt = ii >> 4, jj = ii & 15, kc = dd >> 5, kk = dd & 31;
      Kp[(((size_t)bh * 128 + jt) * 2 + kc) * 512 + b_lane(jj, kk) * 16 + b_half(kk)] = bv;
    } else {                   // V: B-frag column = dd, k = token ii
      int kt = ii >> 5, kk = ii & 31, dt = dd >> 4, cc = dd & 15;
      Vp[(((size_t)bh * 64 + kt) * 4 + dt) * 512 + b_lane(cc, kk) * 16 + b_half(kk)] = bv;
    }
  }
}

// =====================================================================
// Kernel 2: fused attention per (bh, 16-row strip).
// Softmax entirely in the exp2 domain (scale' = d^-1/2 * log2 e), so every
// exponential is a bare v_exp_f32.  K lines for the next iteration and this
// iteration's V lines are prefetched (global_prefetch_b8) to overlap with WMMA.
// =====================================================================
__global__ __launch_bounds__(256) void attn_kernel(
    const v16bf* __restrict__ Qf, const v16bf* __restrict__ Kf,
    const v16bf* __restrict__ Vf,
    float* __restrict__ attn_out, float* __restrict__ A_ws)
{
  const int bh    = blockIdx.y;          // 0..31
  const int strip = blockIdx.x;          // 0..127
  const int i0    = strip * 16;
  const int tid   = threadIdx.x;
  const int wave  = tid >> 5;            // 0..7
  const int lane  = tid & 31;
  const int col   = lane & 15;
  const float sc2 = 0.125f * LOG2E;      // 64^-0.5 * log2(e)

  __shared__ v16bf Pst[8][32];           // per-wave P staging, already A-frag layout
  __shared__ float red_m[8][16], red_s[8][16];
  __shared__ float fin_m[16], fin_rs[16];
  __shared__ float Osh[16][DIM];

  for (int t = tid; t < 16 * DIM; t += 256) ((float*)Osh)[t] = 0.0f;

  // Q strip fragments: one 32B load each
  const size_t qbase = ((size_t)bh * 128 + strip) * 2 * 32;
  v16bf qa0 = Qf[qbase + lane];
  v16bf qa1 = Qf[qbase + 32 + lane];

  // ---- pass 1: online softmax statistics (log2 domain)
  float m[8], s[8];
  for (int v = 0; v < 8; ++v) { m[v] = -3.0e38f; s[v] = 0.0f; }

  for (int t = 0; t < 16; ++t) {
    const int jt = wave * 16 + t;
    const size_t kbase = ((size_t)bh * 128 + jt) * 2 * 32;
    v16bf kb0 = Kf[kbase + lane];
    v16bf kb1 = Kf[kbase + 32 + lane];
    if (t < 15) {  // prefetch next K fragment lines while WMMA runs
      const size_t knext = kbase + 64;
      __builtin_prefetch(Kf + knext + lane, 0, 0);
      __builtin_prefetch(Kf + knext + 32 + lane, 0, 0);
    }
    v8f acc = {};
    acc = WMMA_BF16(qa0, kb0, acc);
    acc = WMMA_BF16(qa1, kb1, acc);
    for (int v = 0; v < 8; ++v) {
      float sv = acc[v] * sc2;
      float nm = fmaxf(m[v], sv);
      s[v] = s[v] * exp2f(m[v] - nm) + exp2f(sv - nm);
      m[v] = nm;
    }
  }
  for (int mask = 1; mask < 16; mask <<= 1)
    for (int v = 0; v < 8; ++v) {
      float om = __shfl_xor(m[v], mask);
      float os = __shfl_xor(s[v], mask);
      float nm = fmaxf(m[v], om);
      s[v] = s[v] * exp2f(m[v] - nm) + os * exp2f(om - nm);
      m[v] = nm;
    }
  if (lane == 0 || lane == 16)
    for (int v = 0; v < 8; ++v) {
      red_m[wave][c_row(lane, v)] = m[v];
      red_s[wave][c_row(lane, v)] = s[v];
    }
  __syncthreads();
  if (tid < 16) {
    float M = -3.0e38f, S = 0.0f;
    for (int w = 0; w < 8; ++w) {
      float wm = red_m[w][tid], ws = red_s[w][tid];
      float nm = fmaxf(M, wm);
      S = S * exp2f(M - nm) + ws * exp2f(wm - nm);
      M = nm;
    }
    fin_m[tid]  = M;
    fin_rs[tid] = rcpf(S);
  }
  __syncthreads();

  float fm[8], fr[8];
  for (int v = 0; v < 8; ++v) { int r = c_row(lane, v); fm[v] = fin_m[r]; fr[v] = fin_rs[r]; }

  // ---- pass 2: recompute, write attn, accumulate O = P @ V
  v8f z = {};
  v8f oacc0 = z, oacc1 = z, oacc2 = z, oacc3 = z;
  float* attn_base = attn_out + (size_t)bh * NN * NN + (size_t)i0 * NN;
  __bf16* pw = (__bf16*)(&Pst[wave][0]);

  for (int p = 0; p < 8; ++p) {
    const int kt = wave * 8 + p;            // 32-wide j chunk
    const int jbase = kt * 32;
    // prefetch this iteration's V fragments; computed well before their use
    const size_t vbase = ((size_t)bh * 64 + kt) * 4 * 32;
    __builtin_prefetch(Vf + vbase + lane, 0, 0);
    __builtin_prefetch(Vf + vbase + 64 + lane, 0, 0);
    for (int sub = 0; sub < 2; ++sub) {
      const int j0 = jbase + sub * 16;
      const int jt = j0 >> 4;
      const size_t kbase = ((size_t)bh * 128 + jt) * 2 * 32;
      v16bf kb0 = Kf[kbase + lane];
      v16bf kb1 = Kf[kbase + 32 + lane];
      v8f acc = {};
      acc = WMMA_BF16(qa0, kb0, acc);
      acc = WMMA_BF16(qa1, kb1, acc);
      const int jl = sub * 16 + col;        // local k index 0..31 within chunk
      const int lp16 = (jl & 8) ? 16 : 0;
      const int hp = a_half(jl);
      for (int v = 0; v < 8; ++v) {
        const int r = c_row(lane, v);
        const float pv = exp2f(acc[v] * sc2 - fm[v]) * fr[v];
        attn_base[(size_t)r * NN + j0 + col] = pv;
        pw[(r + lp16) * 16 + hp] = f2bf(pv);
      }
    }
    __syncthreads();
    v16bf pa = Pst[wave][lane];
    oacc0 = WMMA_BF16(pa, Vf[vbase + lane],          oacc0);
    oacc1 = WMMA_BF16(pa, Vf[vbase + 32 + lane],     oacc1);
    oacc2 = WMMA_BF16(pa, Vf[vbase + 64 + lane],     oacc2);
    oacc3 = WMMA_BF16(pa, Vf[vbase + 96 + lane],     oacc3);
    __syncthreads();
  }

  for (int v = 0; v < 8; ++v) {
    const int r = c_row(lane, v);
    atomicAdd(&Osh[r][col],      oacc0[v]);
    atomicAdd(&Osh[r][16 + col], oacc1[v]);
    atomicAdd(&Osh[r][32 + col], oacc2[v]);
    atomicAdd(&Osh[r][48 + col], oacc3[v]);
  }
  __syncthreads();

  const int b = bh >> 3, hh = bh & 7;
  for (int t = tid; t < 16 * DIM; t += 256) {
    int r = t >> 6, dd = t & 63;
    A_ws[((size_t)(b * NN + i0 + r)) * INNER + hh * DIM + dd] = Osh[r][dd];
  }
}

// =====================================================================
// Kernel 3: LayerNorm over INNER=512, writing bf16 directly in A-frag
// layout for the projection GEMM:  A1p[strip(512)][kc(16)][lane][h].
// =====================================================================
__global__ __launch_bounds__(256) void ln1_kernel(
    const float* __restrict__ A_ws, const float* __restrict__ g1,
    const float* __restrict__ be1, __bf16* __restrict__ A1p)
{
  const int strip = blockIdx.x;           // 0..511
  const int tid = threadIdx.x;
  const int r   = tid >> 4;               // row in strip
  const int kc  = tid & 15;               // 32-wide k chunk
  const float* x = A_ws + ((size_t)strip * 16 + r) * INNER + kc * 32;

  float vbuf[32];
  float s = 0.0f, s2 = 0.0f;
  for (int j = 0; j < 32; j += 4) {
    float4 v4 = *(const float4*)(x + j);
    vbuf[j] = v4.x; vbuf[j + 1] = v4.y; vbuf[j + 2] = v4.z; vbuf[j + 3] = v4.w;
    s  += v4.x + v4.y + v4.z + v4.w;
    s2 += v4.x * v4.x + v4.y * v4.y + v4.z * v4.z + v4.w * v4.w;
  }
  for (int mask = 1; mask < 16; mask <<= 1) {
    s  += __shfl_xor(s, mask);
    s2 += __shfl_xor(s2, mask);
  }
  const float mu   = s * (1.0f / INNER);
  const float var  = s2 * (1.0f / INNER) - mu * mu;
  const float rstd = rsqrtf(var + EPS);

  __bf16* dst = A1p + ((size_t)strip * 16 + kc) * 512;   // [strip][kc] block
  for (int kl = 0; kl < 32; ++kl) {
    const int k = kc * 32 + kl;
    const float y = (vbuf[kl] - mu) * rstd * g1[k] + be1[k];
    dst[a_lane(r, kl) * 16 + a_half(kl)] = f2bf(y);
  }
}

// =====================================================================
// Kernel 4: proj GEMM (WMMA, K=512, fragments are single 32B loads)
// + bias + LayerNorm(64) + relu residual.
// =====================================================================
__global__ __launch_bounds__(128) void proj_kernel(
    const v16bf* __restrict__ A1p, const v16bf* __restrict__ WjP,
    const float* __restrict__ bp,  const float* __restrict__ g2,
    const float* __restrict__ be2, const float* __restrict__ cur,
    float* __restrict__ out)
{
  const int strip = blockIdx.x;
  const int m0   = strip * 16;
  const int tid  = threadIdx.x;
  const int wave = tid >> 5;   // 0..3 == column tile
  const int lane = tid & 31;
  const int col  = lane & 15;

  __shared__ float Ysh[16][DIM];

  v8f acc = {};
  for (int kc = 0; kc < 16; ++kc) {
    v16bf af = A1p[((size_t)strip * 16 + kc) * 32 + lane];
    v16bf bf = WjP[((size_t)kc * 4 + wave) * 32 + lane];
    acc = WMMA_BF16(af, bf, acc);
  }
  const int cidx = wave * 16 + col;
  const float bias = bp[cidx];
  for (int v = 0; v < 8; ++v) Ysh[c_row(lane, v)][cidx] = acc[v] + bias;
  __syncthreads();

  // LN over 64 + relu residual: 8 threads per row
  const int r  = tid >> 3;
  const int t8 = tid & 7;
  float vals[8];
  float s = 0.0f, s2 = 0.0f;
  for (int j = 0; j < 8; ++j) {
    float v = Ysh[r][t8 * 8 + j];
    vals[j] = v; s += v; s2 += v * v;
  }
  for (int mask = 1; mask < 8; mask <<= 1) {
    s  += __shfl_xor(s, mask);
    s2 += __shfl_xor(s2, mask);
  }
  const float mu   = s * (1.0f / DIM);
  const float var  = s2 * (1.0f / DIM) - mu * mu;
  const float rstd = rsqrtf(var + EPS);
  const int grow = m0 + r;
  const float* cr = cur + (size_t)grow * DIM + t8 * 8;
  float ov[8];
  for (int j = 0; j < 8; ++j) {
    const int d = t8 * 8 + j;
    float y = (vals[j] - mu) * rstd * g2[d] + be2[d];
    ov[j] = y + fmaxf(cr[j], 0.0f);
  }
  float4 o0 = { ov[0], ov[1], ov[2], ov[3] };
  float4 o1 = { ov[4], ov[5], ov[6], ov[7] };
  float4* outv = (float4*)(out + (size_t)grow * DIM + t8 * 8);
  outv[0] = o0;
  outv[1] = o1;
}

// =====================================================================
extern "C" void kernel_launch(void* const* d_in, const int* in_sizes, int n_in,
                              void* d_out, int out_size, void* d_ws, size_t ws_size,
                              hipStream_t stream) {
  (void)in_sizes; (void)n_in; (void)out_size; (void)ws_size;
  const float* cur   = (const float*)d_in[0];
  const float* pre   = (const float*)d_in[1];
  const float* Wc    = (const float*)d_in[2];
  const float* bc    = (const float*)d_in[3];
  const float* Wp    = (const float*)d_in[4];
  const float* bpre  = (const float*)d_in[5];
  const float* alpha = (const float*)d_in[6];
  const float* Wproj = (const float*)d_in[7];
  const float* bproj = (const float*)d_in[8];
  const float* g1    = (const float*)d_in[9];
  const float* be1   = (const float*)d_in[10];
  const float* g2    = (const float*)d_in[11];
  const float* be2   = (const float*)d_in[12];

  float* outA    = (float*)d_out;                         // [4,2048,64]
  float* outAttn = (float*)d_out + (size_t)TOTROW * DIM;  // [4,8,2048,2048]

  // workspace layout
  char* ws = (char*)d_ws;
  const size_t MB = 1024 * 1024;
  __bf16* Qp  = (__bf16*)(ws);              //  8 MB, packed A-frags
  __bf16* Kp  = (__bf16*)(ws + 8 * MB);     //  8 MB, packed B-frags
  __bf16* Vp  = (__bf16*)(ws + 16 * MB);    //  8 MB, packed B-frags
  float*  Aws = (float*) (ws + 24 * MB);    // 16 MB
  __bf16* A1p = (__bf16*)(ws + 40 * MB);    //  8 MB, packed A-frags
  __bf16* WcP = (__bf16*)(ws + 48 * MB);    // 192 KB
  __bf16* WpP = (__bf16*)(ws + 48 * MB + 256 * 1024);
  __bf16* WjP = (__bf16*)(ws + 48 * MB + 512 * 1024);

  // 0) pack weights into fragment layout
  pack_w_kernel<<<dim3(384), dim3(256), 0, stream>>>(Wc, Wp, Wproj, WcP, WpP, WjP);

  // 1) gated QKV -> pre-swizzled Q/K/V fragments
  qkv_gate_kernel<<<dim3(6144), dim3(256), 0, stream>>>(
      cur, pre, (const v16bf*)WcP, (const v16bf*)WpP, bc, bpre, alpha, Qp, Kp, Vp);

  // 2) fused attention
  attn_kernel<<<dim3(NN / 16, BB * HEADS), dim3(256), 0, stream>>>(
      (const v16bf*)Qp, (const v16bf*)Kp, (const v16bf*)Vp, outAttn, Aws);

  // 3) layernorm over INNER -> packed A-frags
  ln1_kernel<<<dim3(TOTROW / 16), dim3(256), 0, stream>>>(Aws, g1, be1, A1p);

  // 4) projection + LN2 + relu residual
  proj_kernel<<<dim3(TOTROW / 16), dim3(128), 0, stream>>>(
      (const v16bf*)A1p, (const v16bf*)WjP, bproj, g2, be2, cur, outA);
}
